// self_attention_67095979098140
// MI455X (gfx1250) — compile-verified, matrix operand order checked by reading, not run on hardware
//
#include <hip/hip_runtime.h>
#include <hip/hip_bf16.h>

// ---------------------------------------------------------------------------
// Self-attention (B=8, C=512, N=2048) for gfx1250 (MI455X, wave32, WMMA).
//   Kernel 1: fused QKV projection, f16 WMMA, f32 accum; outputs written
//             pre-swizzled into WMMA operand-fragment layout.
//   Kernel 2: flash attention.  Per 256-key block-iteration each wave
//             computes S for ITS OWN 32-key chunk (no duplicated WMMA),
//             chunk softmax stats + P fragments are shared through
//             double-buffered LDS, then every wave runs PV for its 64
//             output channels over all 8 shared P fragments.  Next-block
//             K/V fragments are prefetched (global_prefetch_b8) while the
//             current block's WMMAs execute.
// ---------------------------------------------------------------------------

typedef __attribute__((ext_vector_type(16))) _Float16 v16h;
typedef __attribute__((ext_vector_type(8)))  float    v8f;

#define B_ 8
#define C_ 512
#define H_ 256     // C/2, Q/K channels
#define N_ 2048

// K-index striping for 16-bit A/B operands of v_wmma_f32_16x16x32_f16
// (ISA 7.12.2): element e of the per-lane v16h maps to
//   e < 8 : K = 8*half + e       ;  e >= 8: K = 16 + 8*half + (e - 8)
// Branchless inverse (K -> (half, e)):
__device__ __forceinline__ void kinv(int k, int& hh, int& ee) {
  hh = (k >> 3) & 1;
  ee = (k & 7) | ((k >> 1) & 8);
}

__device__ __forceinline__ v8f wmma16(v16h a, v16h b, v8f c) {
  return __builtin_amdgcn_wmma_f32_16x16x32_f16(false, a, false, b, (short)0, c,
                                                false, false);
}

// Fragment workspace (f16; 512 elements = one 16x32 operand tile stored as
// [lane][e] so each lane reads its 16 values with one 32-byte load):
//   fq[b][qt][kc8] : A-operand, row = query, K = head channel
//   fk[b][mt][kc8] : B-operand, col = key,   K = head channel
//   fv[b][km][ct]  : B-operand, col = value channel, K = key within chunk
#define FQ_ELEMS ((size_t)B_ * 128 * 8 * 512)
#define FK_ELEMS ((size_t)B_ * 128 * 8 * 512)
#define FV_ELEMS ((size_t)B_ * 64 * 32 * 512)

// ---------------------------------------------------------------------------
// Kernel 1: QKV projection (one wave = one 16x16 output tile).
// ---------------------------------------------------------------------------
__global__ __launch_bounds__(256) void qkv_proj_kernel(
    const float* __restrict__ x,
    const float* __restrict__ Wq, const float* __restrict__ bq,
    const float* __restrict__ Wk, const float* __restrict__ bk,
    const float* __restrict__ Wv, const float* __restrict__ bv,
    _Float16* __restrict__ fq, _Float16* __restrict__ fk,
    _Float16* __restrict__ fv) {
  __shared__ float xtile[8][512];   // per-wave 32(k) x 16(n) f32 staging

  const int lane = threadIdx.x & 31;
  const int wave = threadIdx.x >> 5;
  const int half = lane >> 4;
  const int lo   = lane & 15;

  int tile = blockIdx.x * 8 + wave;   // [0, 65536)
  int b    = tile >> 13;
  int r2   = tile & 8191;
  int o0   = (r2 >> 7) << 4;
  int n0   = (r2 & 127) << 4;

  const float* W; const float* bias; int mode;
  if (o0 < 256)      { mode = 0; W = Wq + (size_t)o0 * C_;         bias = bq + o0; }
  else if (o0 < 512) { mode = 1; W = Wk + (size_t)(o0 - 256) * C_; bias = bk + (o0 - 256); }
  else               { mode = 2; W = Wv + (size_t)(o0 - 512) * C_; bias = bv + (o0 - 512); }

  const float* xb = x + (size_t)b * C_ * N_;

  v8f acc = {0.f, 0.f, 0.f, 0.f, 0.f, 0.f, 0.f, 0.f};
  for (int kc = 0; kc < C_; kc += 32) {
#pragma unroll
    for (int t = 0; t < 4; ++t) {
      int idx = lane + 32 * t;
      int row = idx >> 2;
      int cg  = (idx & 3) << 2;
      *(float4*)&xtile[wave][row * 16 + cg] =
          *(const float4*)&xb[(size_t)(kc + row) * N_ + n0 + cg];
    }

    const float* wrow = W + (size_t)lo * C_ + kc;
    float4 w0 = *(const float4*)(wrow + 8 * half);
    float4 w1 = *(const float4*)(wrow + 8 * half + 4);
    float4 w2 = *(const float4*)(wrow + 16 + 8 * half);
    float4 w3 = *(const float4*)(wrow + 16 + 8 * half + 4);
    v16h a;
    a[0]=(_Float16)w0.x;  a[1]=(_Float16)w0.y;  a[2]=(_Float16)w0.z;  a[3]=(_Float16)w0.w;
    a[4]=(_Float16)w1.x;  a[5]=(_Float16)w1.y;  a[6]=(_Float16)w1.z;  a[7]=(_Float16)w1.w;
    a[8]=(_Float16)w2.x;  a[9]=(_Float16)w2.y;  a[10]=(_Float16)w2.z; a[11]=(_Float16)w2.w;
    a[12]=(_Float16)w3.x; a[13]=(_Float16)w3.y; a[14]=(_Float16)w3.z; a[15]=(_Float16)w3.w;

    v16h bm;
#pragma unroll
    for (int e = 0; e < 16; ++e) {
      int k = (e < 8) ? (8 * half + e) : (16 + 8 * half + (e - 8));
      bm[e] = (_Float16)xtile[wave][k * 16 + lo];
    }
    acc = wmma16(a, bm, acc);
  }

  _Float16 vals[8];
#pragma unroll
  for (int r = 0; r < 8; ++r)
    vals[r] = (_Float16)(acc[r] + bias[r + 8 * half]);

  if (mode < 2) {
    int oc    = (mode == 0) ? o0 : (o0 - 256);
    int kc8   = oc >> 5;
    int kbase = oc & 31;
    _Float16* frag = ((mode == 0) ? fq : fk) +
                     (((size_t)b * 128 + (n0 >> 4)) * 8 + kc8) * 512;
#pragma unroll
    for (int r = 0; r < 8; ++r) {
      int kl = kbase + r + 8 * half;
      int hh, ee; kinv(kl, hh, ee);
      frag[(hh * 16 + lo) * 16 + ee] = vals[r];
    }
  } else {
    int ct    = (o0 - 512) >> 4;
    int kbase = n0 & 31;
    _Float16* frag = fv + (((size_t)b * 64 + (n0 >> 5)) * 32 + ct) * 512;
    int kl = kbase + lo;
    int hh, ee; kinv(kl, hh, ee);
#pragma unroll
    for (int r = 0; r < 8; ++r)
      frag[(hh * 16 + (r + 8 * half)) * 16 + ee] = vals[r];
  }
}

// ---------------------------------------------------------------------------
// Kernel 2: flash attention, cooperative S (no duplicated WMMA).
// Grid: B * (N/16) = 1024 blocks x 256 threads (8 waves).
// ---------------------------------------------------------------------------
__global__ __launch_bounds__(256) void flash_attn_kernel(
    const _Float16* __restrict__ fq, const _Float16* __restrict__ fk,
    const _Float16* __restrict__ fv, const float* __restrict__ x,
    float* __restrict__ out) {
  __shared__ __align__(32) _Float16 pLDS[2][8][512];   // [buf][chunk][frag]
  __shared__ __align__(32) float    mxLDS[2][16][8];   // [buf][row][chunk]
  __shared__ __align__(32) float    smLDS[2][16][8];   // [buf][row][chunk]

  const int lane = threadIdx.x & 31;
  const int wave = threadIdx.x >> 5;
  const int half = lane >> 4;
  const int lo   = lane & 15;

  const int b  = blockIdx.x >> 7;
  const int qt = blockIdx.x & 127;
  const int n0 = qt << 4;
  const int c0 = wave * 64;

  // Preload Q tile: 8 A-operand fragments (one 32B load each).
  const _Float16* fqb = fq + ((size_t)b * 128 + qt) * 8 * 512;
  v16h qa[8];
#pragma unroll
  for (int kc8 = 0; kc8 < 8; ++kc8)
    qa[kc8] = *(const v16h*)(fqb + kc8 * 512 + lane * 16);

  v8f zero = {0.f, 0.f, 0.f, 0.f, 0.f, 0.f, 0.f, 0.f};
  v8f acc[4] = {zero, zero, zero, zero};
  float m_row[8], l_row[8];
#pragma unroll
  for (int r = 0; r < 8; ++r) { m_row[r] = -1e30f; l_row[r] = 0.f; }

  const float scale = 0.0625f;  // 1/sqrt(256)

  // P-fragment LDS write offsets:
  //   col0 = lo      -> (hh = lo>>3, ee = lo&7)
  //   col1 = 16 + lo -> (hh = lo>>3, ee = (lo&7)|8)
  const int h0 = lo >> 3;
  const int e0 = lo & 7;
  const int e1 = (lo & 7) | 8;

  int it = 0;
  for (int kb = 0; kb < N_; kb += 256, ++it) {
    const int buf = it & 1;
    const int m0  = kb + wave * 32;   // this wave's key chunk

    // ---- prefetch next block's K/V fragments into near caches ----
    // (global_prefetch_b8; everything is L2-resident, this pulls the next
    //  256-key working set toward the WGP while this block's WMMAs run.)
    if (kb + 256 < N_) {
      const char* pfk = (const char*)(fk +
          ((size_t)b * 128 + ((kb + 256 + wave * 32) >> 4)) * 8 * 512);
      __builtin_prefetch(pfk + lane * 256, 0, 3);     // 32 x 256B = 8KB
#pragma unroll
      for (int w2 = 0; w2 < 8; ++w2) {
        const char* pfv = (const char*)(fv +
            (((size_t)b * 64 + ((kb + 256) >> 5) + w2) * 32 + (c0 >> 4)) * 512);
        __builtin_prefetch(pfv + lane * 128, 0, 3);   // 32 x 128B = 4KB
      }
    }

    // ---- phase A: S for own chunk (16 WMMA, no duplication) ----
    const _Float16* fkb0 = fk + ((size_t)b * 128 + (m0 >> 4)) * 8 * 512;
    const _Float16* fkb1 = fkb0 + 8 * 512;
    v8f s0 = zero, s1 = zero;
#pragma unroll
    for (int kc8 = 0; kc8 < 8; ++kc8) {
      v16h kb0v = *(const v16h*)(fkb0 + kc8 * 512 + lane * 16);
      v16h kb1v = *(const v16h*)(fkb1 + kc8 * 512 + lane * 16);
      s0 = wmma16(qa[kc8], kb0v, s0);
      s1 = wmma16(qa[kc8], kb1v, s1);
    }

    float ps0[8], ps1[8];
#pragma unroll
    for (int r = 0; r < 8; ++r) {
      ps0[r] = s0[r] * scale;
      ps1[r] = s1[r] * scale;
      float t = fmaxf(ps0[r], ps1[r]);
#pragma unroll
      for (int off = 8; off > 0; off >>= 1) t = fmaxf(t, __shfl_xor(t, off, 16));
      if (lo == 0) mxLDS[buf][r + 8 * half][wave] = t;   // lanes 0 and 16
    }
    __syncthreads();

    // ---- phase B: merge maxes, exponentiate own chunk, publish P ----
    float corr[8];
#pragma unroll
    for (int r = 0; r < 8; ++r) {
      int row = r + 8 * half;
      v8f mv = *(const v8f*)&mxLDS[buf][row][0];
      float bmx = mv[0];
#pragma unroll
      for (int w = 1; w < 8; ++w) bmx = fmaxf(bmx, mv[w]);
      float mn = fmaxf(m_row[r], bmx);
      corr[r]  = __expf(m_row[r] - mn);
      m_row[r] = mn;

      float p0 = __expf(ps0[r] - mn);
      float p1 = __expf(ps1[r] - mn);
      float rs = p0 + p1;
#pragma unroll
      for (int off = 8; off > 0; off >>= 1) rs += __shfl_xor(rs, off, 16);
      if (lo == 0) smLDS[buf][row][wave] = rs;
      pLDS[buf][wave][h0 * 256 + row * 16 + e0] = (_Float16)p0;
      pLDS[buf][wave][h0 * 256 + row * 16 + e1] = (_Float16)p1;
    }
    __syncthreads();

    // ---- phase C: merge sums, rescale, PV over all 8 chunks ----
#pragma unroll
    for (int r = 0; r < 8; ++r) {
      int row = r + 8 * half;
      v8f sv = *(const v8f*)&smLDS[buf][row][0];
      float ssum = sv[0] + sv[1] + sv[2] + sv[3] + sv[4] + sv[5] + sv[6] + sv[7];
      l_row[r] = l_row[r] * corr[r] + ssum;
      acc[0][r] *= corr[r]; acc[1][r] *= corr[r];
      acc[2][r] *= corr[r]; acc[3][r] *= corr[r];
    }

    const _Float16* fvb_base =
        fv + (((size_t)b * 64 + (kb >> 5)) * 32 + (c0 >> 4)) * 512;
#pragma unroll
    for (int w2 = 0; w2 < 8; ++w2) {
      v16h pa = *(const v16h*)&pLDS[buf][w2][lane * 16];
      const _Float16* fvb = fvb_base + (size_t)w2 * 32 * 512;
#pragma unroll
      for (int j = 0; j < 4; ++j) {
        v16h vb = *(const v16h*)(fvb + (size_t)j * 512 + lane * 16);
        acc[j] = wmma16(pa, vb, acc[j]);
      }
    }
    // Next iteration uses the other LDS buffer; the two barriers above
    // order buffer reuse across waves.
  }

  // ---- epilogue: normalize, add residual, vectorized f32 store ----
  float rl[8];
#pragma unroll
  for (int r = 0; r < 8; ++r) rl[r] = 1.0f / l_row[r];

  const float* xb = x   + (size_t)b * C_ * N_;
  float*       ob = out + (size_t)b * C_ * N_;
#pragma unroll
  for (int j = 0; j < 4; ++j) {
    int c = c0 + j * 16 + lo;
    size_t base = (size_t)c * N_ + n0 + 8 * half;
    v8f xv = *(const v8f*)(xb + base);
    v8f res;
#pragma unroll
    for (int r = 0; r < 8; ++r) res[r] = xv[r] + acc[j][r] * rl[r];
    *(v8f*)(ob + base) = res;
  }
}

// ---------------------------------------------------------------------------
extern "C" void kernel_launch(void* const* d_in, const int* in_sizes, int n_in,
                              void* d_out, int out_size, void* d_ws,
                              size_t ws_size, hipStream_t stream) {
  const float* x  = (const float*)d_in[0];
  const float* Wq = (const float*)d_in[1];
  const float* bq = (const float*)d_in[2];
  const float* Wk = (const float*)d_in[3];
  const float* bk = (const float*)d_in[4];
  const float* Wv = (const float*)d_in[5];
  const float* bv = (const float*)d_in[6];
  float* out = (float*)d_out;

  _Float16* fq = (_Float16*)d_ws;
  _Float16* fk = fq + FQ_ELEMS;
  _Float16* fv = fk + FK_ELEMS;

  qkv_proj_kernel<<<8192, 256, 0, stream>>>(x, Wq, bq, Wk, bk, Wv, bv,
                                            fq, fk, fv);
  flash_attn_kernel<<<1024, 256, 0, stream>>>(fq, fk, fv, x, out);
}
